// ResNet50_YOLOv8_29875792511850
// MI455X (gfx1250) — compile-verified, matrix operand order checked by reading, not run on hardware
//
#include <hip/hip_runtime.h>
#include <hip/hip_bf16.h>
#include <math.h>
#include <stdint.h>

// ---------------------------------------------------------------------------
// CDNA5 (gfx1250) wave32 WMMA types
// ---------------------------------------------------------------------------
typedef __bf16  v16bf __attribute__((ext_vector_type(16)));
typedef float   v8f   __attribute__((ext_vector_type(8)));
typedef unsigned short u16;
typedef unsigned int   u32;
typedef unsigned long long u64;

union FragBF { uint4 u[2]; v16bf v; };
union Pack8  { u16 h[8]; uint4 q; };

__device__ __forceinline__ u16 f2bf(float f) {
    // round-to-nearest-even f32 -> bf16
    u32 u = __float_as_uint(f);
    u32 r = u + 0x7FFFu + ((u >> 16) & 1u);
    return (u16)(r >> 16);
}
__device__ __forceinline__ float sigmoidf(float v) { return 1.f / (1.f + __expf(-v)); }
__device__ __forceinline__ float clamp01(float v)  { return fminf(fmaxf(v, 0.f), 1.f); }
// exact unsigned div for n < 2^24, d < 2^16 via magic = (2^40/d)+1
__device__ __forceinline__ u32 mdiv(u32 n, u64 magic) { return (u32)(((u64)n * magic) >> 40); }

#define BM 64
#define BN 128
#define BK 32

// epilogue shared by both conv kernels
__device__ __forceinline__ void conv_epilogue(
    const v8f* acc, const float* scale, const float* bias, const float* res,
    float* Y, int m0, int n0, int mt, int nt0, int lane,
    int Cout, int Nsp, int bz, int act, float bnmul)
{
#pragma unroll
    for (int q = 0; q < 4; ++q) {
        v8f cc = acc[q];
        int nGl = n0 + (nt0 + q) * 16 + (lane & 15);
#pragma unroll
        for (int r = 0; r < 8; ++r) {
            int mGl = m0 + mt * 16 + r + 8 * (lane >> 4);
            if (mGl < Cout && nGl < Nsp) {
                float v = cc[r];
                float s = scale ? scale[mGl] * bnmul : 1.f;
                float b = bias  ? bias[mGl] : 0.f;
                v = v * s + b;
                size_t oidx = ((size_t)bz * Cout + mGl) * (size_t)Nsp + nGl;
                if (res) v += res[oidx];
                if      (act == 1) v = fmaxf(v, 0.f);
                else if (act == 2) v = (v > 0.f) ? v : 0.1f * v;
                else if (act == 3) v = sigmoidf(v);
                Y[oidx] = v;
            }
        }
    }
}

// ---------------------------------------------------------------------------
// Weight repack: f32 OIHW -> bf16 fragment-ordered 4KB tiles.
// Tile (mblk, khkw, cc) holds the 64x32 (M x ci) fragment-ordered block:
//   byte layout matches the conv kernel's LDS A-tile exactly, so the conv
//   A-stage is a contiguous 4KB copy (async-to-LDS).
// Tile index = (mblk*KHKW + khkw)*CinT + cc   (CinT = Cin/32)
// grid = (CinT, KHKW, nMB), 256 threads (8 u16 per thread).
// ---------------------------------------------------------------------------
__global__ void repack_w_k(const float* __restrict__ Wt, u16* __restrict__ Wp,
                           int Cout, int Cin, int KHKW)
{
    const int cc   = blockIdx.x;
    const int khkw = blockIdx.y;
    const int mblk = blockIdx.z;
    const int CinT = Cin >> 5;
    const int Ktot = Cin * KHKW;
    u16* dst = Wp + (((size_t)mblk * gridDim.y + khkw) * CinT + cc) * 2048;

    const int t = threadIdx.x;
    const int rowA = t >> 2, kcA = t & 3;
    const int m = mblk * 64 + rowA;
    const int aBase = (rowA >> 4) * 512 + ((rowA & 15) + 16 * (kcA & 1)) * 16
                    + (kcA >> 1) * 8;
    Pack8 tmp;
#pragma unroll
    for (int j = 0; j < 8; ++j) {
        int ci = cc * 32 + kcA * 8 + j;
        float v = (m < Cout) ? Wt[(size_t)m * Ktot + (size_t)ci * KHKW + khkw] : 0.f;
        tmp.h[j] = f2bf(v);
    }
    *(uint4*)&dst[aBase] = tmp.q;
}

// ---------------------------------------------------------------------------
// Structured implicit-GEMM conv (requires Cin % 32 == 0), packed bf16 weights.
// K walked as (kh,kw) uniform outer loops x ci chunks of 32.
//  - A tile: contiguous 4KB of pre-swizzled bf16, loaded with
//    global_load_async_to_lds_b128 (ASYNCcnt) straight into LDS.
//  - B tile: im2col gather with per-(kh,kw) hoisted bounds check; addresses
//    are base + immediate offsets.
// Tile: 64(M) x 128(N) x 32(K); 8 waves, 4 WMMA per wave per K step.
// ---------------------------------------------------------------------------
__global__ __launch_bounds__(256, 2)
void conv_wmma_s(const float* __restrict__ X, const u16* __restrict__ Wp,
                 const float* __restrict__ scale, const float* __restrict__ bias,
                 const float* __restrict__ res, float* __restrict__ Y,
                 int Cin, int Cout, int Hin, int Win, int Hout, int Wout,
                 int KH, int KW, int stride, int pad, int act,
                 float bnmul, u64 magicW)
{
    const int Nsp    = Hout * Wout;
    const int HinWin = Hin * Win;
    const int KHKW   = KH * KW;
    const int CinT   = Cin >> 5;
    const int n0 = blockIdx.x * BN;
    const int m0 = blockIdx.y * BM;
    const int bz = blockIdx.z;
    const int t    = threadIdx.x;
    const int lane = t & 31;
    const int wv   = t >> 5;

    __shared__ alignas(16) u16 lA[4 * 32 * 16];  // 4 KB A tile (fragment order)
    __shared__ alignas(16) u16 lB[8 * 32 * 16];  // 8 KB B tile (fragment order)

    // per-thread async A copy: 16 bytes each (256 x 16B = 4KB)
    const u32 ldsA = (u32)(uintptr_t)&lA[0] + (u32)t * 16u;
    const u16* wBlk = Wp + (size_t)blockIdx.y * KHKW * CinT * 2048;

    // B loader: 2 chunks of 8 ci for one N column
    const int rowB = t >> 1;
    const int kcB0 = (t & 1) * 2;
    const int nG = n0 + rowB;
    u32 ohB = 0, owB = 0;
    if (nG < Nsp) {
        ohB = mdiv((u32)nG, magicW);
        owB = (u32)nG - ohB * (u32)Wout;
    }
    const float* xImg = X + (size_t)bz * Cin * HinWin;

    const int mt  = wv & 3;
    const int nt0 = (wv >> 2) * 4;

    v8f acc[4] = {};

    for (int kh = 0; kh < KH; ++kh) {
        for (int kw = 0; kw < KW; ++kw) {
            const int khkw = kh * KW + kw;
            const int ih = (int)ohB * stride - pad + kh;
            const int iw = (int)owB * stride - pad + kw;
            const bool bvalid = (nG < Nsp) && ih >= 0 && ih < Hin && iw >= 0 && iw < Win;
            const float* xBase = xImg + ((ptrdiff_t)ih * Win + iw);
            const u16* wTile0 = wBlk + (size_t)khkw * CinT * 2048;

            for (int c0 = 0; c0 < Cin; c0 += BK) {
                // ---- stage A: async copy of the packed 4KB tile into LDS ----
                {
                    u64 ga = (u64)(uintptr_t)(wTile0 + (size_t)(c0 >> 5) * 2048)
                           + (u64)t * 16u;
                    asm volatile("global_load_async_to_lds_b128 %0, %1, off"
                                 :: "v"(ldsA), "v"(ga) : "memory");
                }
                // ---- stage B: 16 gathers, stride HinWin (immediate offsets) ----
#pragma unroll
                for (int c = 0; c < 2; ++c) {
                    const int kc = kcB0 + c;
                    const float* xp = xBase + (size_t)(c0 + kc * 8) * HinWin;
                    Pack8 tmp;
#pragma unroll
                    for (int j = 0; j < 8; ++j)
                        tmp.h[j] = f2bf(bvalid ? xp[(size_t)j * HinWin] : 0.f);
                    const int bBase = (rowB >> 4) * 512
                                    + ((rowB & 15) + 16 * (kc & 1)) * 16 + (kc >> 1) * 8;
                    *(uint4*)&lB[bBase] = tmp.q;
                }
                // drain this wave's async-to-LDS before the tile barrier
                asm volatile("s_wait_asynccnt 0x0" ::: "memory");
                __syncthreads();

                // ---- matrix core: 4 x v_wmma_f32_16x16x32_bf16 per wave ----
                FragBF fa;
                const int aoff = mt * 512 + lane * 16;
                fa.u[0] = *(const uint4*)&lA[aoff];
                fa.u[1] = *(const uint4*)&lA[aoff + 8];
#pragma unroll
                for (int q = 0; q < 4; ++q) {
                    FragBF fb;
                    const int boff = (nt0 + q) * 512 + lane * 16;
                    fb.u[0] = *(const uint4*)&lB[boff];
                    fb.u[1] = *(const uint4*)&lB[boff + 8];
                    acc[q] = __builtin_amdgcn_wmma_f32_16x16x32_bf16(
                        false, fa.v, false, fb.v, (short)0, acc[q], false, false);
                }
                __syncthreads();
            }
        }
    }

    conv_epilogue(acc, scale, bias, res, Y, m0, n0, mt, nt0, lane,
                  Cout, Nsp, bz, act, bnmul);
}

// ---------------------------------------------------------------------------
// Generic implicit-GEMM conv (any Cin/KH/KW; used for the 7x7 stem).
// ---------------------------------------------------------------------------
__global__ __launch_bounds__(256, 2)
void conv_wmma_g(const float* __restrict__ X, const float* __restrict__ Wt,
                 const float* __restrict__ scale, const float* __restrict__ bias,
                 const float* __restrict__ res, float* __restrict__ Y,
                 int Cin, int Cout, int Hin, int Win, int Hout, int Wout,
                 int KH, int KW, int stride, int pad, int act, int clipIn,
                 float bnmul, u64 magicW, u64 magicKHKW, u64 magicKW)
{
    const int Nsp  = Hout * Wout;
    const int KHKW = KH * KW;
    const int Ktot = Cin * KHKW;
    const int n0 = blockIdx.x * BN;
    const int m0 = blockIdx.y * BM;
    const int bz = blockIdx.z;
    const int t    = threadIdx.x;
    const int lane = t & 31;
    const int wv   = t >> 5;

    __shared__ alignas(16) u16 lA[4 * 32 * 16];
    __shared__ alignas(16) u16 lB[8 * 32 * 16];

    const int rowA = t >> 2;
    const int kcA  = t & 3;
    const int aBase = (rowA >> 4) * 512 + ((rowA & 15) + 16 * (kcA & 1)) * 16
                    + (kcA >> 1) * 8;
    const int mG = m0 + rowA;

    const int rowB = t >> 1;
    const int kcB0 = (t & 1) * 2;
    const int nG = n0 + rowB;
    u32 ohB = 0, owB = 0;
    if (nG < Nsp) {
        ohB = mdiv((u32)nG, magicW);
        owB = (u32)nG - ohB * (u32)Wout;
    }
    const size_t xPlane = (size_t)bz * Cin * Hin * Win;

    const int mt  = wv & 3;
    const int nt0 = (wv >> 2) * 4;

    v8f acc[4] = {};

    for (int k0 = 0; k0 < Ktot; k0 += BK) {
        {
            Pack8 tmp;
#pragma unroll
            for (int j = 0; j < 8; ++j) {
                int kGl = k0 + kcA * 8 + j;
                float v = 0.f;
                if (mG < Cout && kGl < Ktot)
                    v = Wt[(size_t)mG * Ktot + kGl];
                tmp.h[j] = f2bf(v);
            }
            *(uint4*)&lA[aBase] = tmp.q;
        }
#pragma unroll
        for (int c = 0; c < 2; ++c) {
            const int kc = kcB0 + c;
            Pack8 tmp;
#pragma unroll
            for (int j = 0; j < 8; ++j) {
                int kGl = k0 + kc * 8 + j;
                float v = 0.f;
                if (nG < Nsp && kGl < Ktot) {
                    u32 ci = mdiv((u32)kGl, magicKHKW);
                    u32 r  = (u32)kGl - ci * (u32)KHKW;
                    u32 kh = mdiv(r, magicKW);
                    u32 kw = r - kh * (u32)KW;
                    int ih = (int)ohB * stride - pad + (int)kh;
                    int iw = (int)owB * stride - pad + (int)kw;
                    if (ih >= 0 && ih < Hin && iw >= 0 && iw < Win) {
                        v = X[xPlane + ((size_t)ci * Hin + ih) * Win + iw];
                        if (clipIn) v = fminf(fmaxf(v, -10.f), 10.f);
                    }
                }
                tmp.h[j] = f2bf(v);
            }
            const int bBase = (rowB >> 4) * 512 + ((rowB & 15) + 16 * (kc & 1)) * 16
                            + (kc >> 1) * 8;
            *(uint4*)&lB[bBase] = tmp.q;
        }
        __syncthreads();

        FragBF fa;
        const int aoff = mt * 512 + lane * 16;
        fa.u[0] = *(const uint4*)&lA[aoff];
        fa.u[1] = *(const uint4*)&lA[aoff + 8];
#pragma unroll
        for (int q = 0; q < 4; ++q) {
            FragBF fb;
            const int boff = (nt0 + q) * 512 + lane * 16;
            fb.u[0] = *(const uint4*)&lB[boff];
            fb.u[1] = *(const uint4*)&lB[boff + 8];
            acc[q] = __builtin_amdgcn_wmma_f32_16x16x32_bf16(
                false, fa.v, false, fb.v, (short)0, acc[q], false, false);
        }
        __syncthreads();
    }

    conv_epilogue(acc, scale, bias, res, Y, m0, n0, mt, nt0, lane,
                  Cout, Nsp, bz, act, bnmul);
}

// ---------------------------------------------------------------------------
// Aux kernels
// ---------------------------------------------------------------------------
__global__ void maxpool_k(const float* __restrict__ in, float* __restrict__ out,
                          int BC, int Hin, int Win, int Hout, int Wout)
{
    int idx = blockIdx.x * blockDim.x + threadIdx.x;
    int total = BC * Hout * Wout;
    if (idx >= total) return;
    int owp = idx % Wout; int t = idx / Wout;
    int ohp = t % Hout;   int bc = t / Hout;
    float m = -INFINITY;
#pragma unroll
    for (int kh = 0; kh < 3; ++kh)
#pragma unroll
        for (int kw = 0; kw < 3; ++kw) {
            int ih = ohp * 2 - 1 + kh, iw = owp * 2 - 1 + kw;
            if (ih >= 0 && ih < Hin && iw >= 0 && iw < Win)
                m = fmaxf(m, in[((size_t)bc * Hin + ih) * Win + iw]);
        }
    out[idx] = m;
}

__global__ void mul_k(const float* __restrict__ a, const float* __restrict__ b,
                      float* __restrict__ y, size_t n)
{
    size_t i = (size_t)blockIdx.x * blockDim.x + threadIdx.x;
    if (i < n) y[i] = a[i] * b[i];
}

__global__ void upsample2_k(const float* __restrict__ in, float* __restrict__ out,
                            int BC, int H, int W)
{
    int idx = blockIdx.x * blockDim.x + threadIdx.x;
    int total = BC * H * 2 * W * 2;
    if (idx >= total) return;
    int W2 = W * 2;
    int ow = idx % W2; int t = idx / W2;
    int oh = t % (H * 2); int bc = t / (H * 2);
    out[idx] = in[((size_t)bc * H + (oh >> 1)) * W + (ow >> 1)];
}

__global__ void decode_k(const float* __restrict__ h3,
                         float* __restrict__ boxes, float* __restrict__ obj,
                         float* __restrict__ lab, int B, int C, int HW)
{
    int idx = blockIdx.x * blockDim.x + threadIdx.x;
    if (idx >= B * HW) return;
    int b = idx / HW, n = idx % HW;
    const float* base = h3 + (size_t)b * C * HW + n;
    float cx = sigmoidf(base[0 * HW]);
    float cy = sigmoidf(base[1 * HW]);
    float w  = sigmoidf(base[2 * HW]);
    float hh = sigmoidf(base[3 * HW]);
    float o  = sigmoidf(base[4 * HW]);
    int   best = 0; float bv = base[5 * HW];
    for (int c = 1; c < C - 5; ++c) {
        float v = base[(5 + c) * HW];
        if (v > bv) { bv = v; best = c; }
    }
    boxes[(size_t)idx * 4 + 0] = clamp01(cx - w * 0.5f);
    boxes[(size_t)idx * 4 + 1] = clamp01(cy - hh * 0.5f);
    boxes[(size_t)idx * 4 + 2] = clamp01(cx + w * 0.5f);
    boxes[(size_t)idx * 4 + 3] = clamp01(cy + hh * 0.5f);
    obj[idx] = o;
    lab[idx] = (float)best;
}

#define NMS_MAXN 1024
#define NMS_EPS  1e-7f
__global__ void nms_k(const float* __restrict__ boxes, const float* __restrict__ scores,
                      float* __restrict__ keepOut, int N, float thr)
{
    int b = blockIdx.x, tid = threadIdx.x;
    __shared__ float sx1[NMS_MAXN], sy1[NMS_MAXN], sx2[NMS_MAXN], sy2[NMS_MAXN], ss[NMS_MAXN];
    __shared__ unsigned char sact[NMS_MAXN], skeep[NMS_MAXN];
    __shared__ float rv[256];
    __shared__ int   ri[256];
    __shared__ int   sel, selValid;

    for (int j = tid; j < N; j += blockDim.x) {
        const float* bp = boxes + ((size_t)b * N + j) * 4;
        sx1[j] = bp[0]; sy1[j] = bp[1]; sx2[j] = bp[2]; sy2[j] = bp[3];
        ss[j] = scores[(size_t)b * N + j];
        sact[j] = 1; skeep[j] = 0;
    }
    __syncthreads();

    for (int it = 0; it < N; ++it) {
        float bv = -INFINITY; int bi = N;
        for (int j = tid; j < N; j += blockDim.x) {
            float v = sact[j] ? ss[j] : -INFINITY;
            if (v > bv || (v == bv && j < bi)) { bv = v; bi = j; }
        }
        rv[tid] = bv; ri[tid] = bi;
        __syncthreads();
        for (int s = 128; s > 0; s >>= 1) {
            if (tid < s) {
                if (rv[tid + s] > rv[tid] ||
                    (rv[tid + s] == rv[tid] && ri[tid + s] < ri[tid])) {
                    rv[tid] = rv[tid + s]; ri[tid] = ri[tid + s];
                }
            }
            __syncthreads();
        }
        if (tid == 0) {
            int i = ri[0]; if (i >= N) i = 0;
            sel = i; selValid = sact[i];
            if (selValid) skeep[i] = 1;
        }
        __syncthreads();
        int i = sel, valid = selValid;
        if (valid) {
            float x1 = sx1[i], y1 = sy1[i], x2 = sx2[i], y2 = sy2[i];
            float area1 = (x2 - x1) * (y2 - y1);
            for (int j = tid; j < N; j += blockDim.x) {
                float x1g = sx1[j], y1g = sy1[j], x2g = sx2[j], y2g = sy2[j];
                float area2 = (x2g - x1g) * (y2g - y1g);
                float iw = fmaxf(fminf(x2, x2g) - fmaxf(x1, x1g), 0.f);
                float ih = fmaxf(fminf(y2, y2g) - fmaxf(y1, y1g), 0.f);
                float inter = iw * ih;
                float iou = inter / (area1 + area2 - inter);
                float uni  = area1 + area2 - inter + NMS_EPS;
                float iou2 = inter / uni;
                float cdx = (x1 + x2) * 0.5f - (x1g + x2g) * 0.5f;
                float cdy = (y1 + y2) * 0.5f - (y1g + y2g) * 0.5f;
                float cd = cdx * cdx + cdy * cdy;
                float edx = fmaxf(x2, x2g) - fminf(x1, x1g);
                float edy = fmaxf(y2, y2g) - fminf(y1, y1g);
                float ed = edx * edx + edy * edy + NMS_EPS;
                float w1 = x2 - x1,  h1 = y2 - y1;
                float w2 = x2g - x1g, h2 = y2g - y1g;
                float da = atanf(w2 / (h2 + NMS_EPS)) - atanf(w1 / (h1 + NMS_EPS));
                float v  = (4.f / (float)(M_PI * M_PI)) * da * da;
                float alpha = v / (1.f - iou2 + v + NMS_EPS);
                float dl = 1.f - (iou2 - cd / ed - alpha * v);
                unsigned char na = (sact[j] && (iou < thr) && (dl > thr) && (j != i)) ? 1 : 0;
                sact[j] = na;
            }
        }
        __syncthreads();
    }
    for (int j = tid; j < N; j += blockDim.x)
        keepOut[(size_t)b * N + j] = skeep[j] ? 1.f : 0.f;
}

// ---------------------------------------------------------------------------
// Host orchestration
// ---------------------------------------------------------------------------
static const float kBNMul = 0.99999500003750f; // 1/sqrt(1+1e-5)

static inline u64 magic40(int d) { return (d > 0) ? ((1ull << 40) / (u64)d) + 1ull : 1ull; }

static inline void launch_conv(hipStream_t st, const float* X, const float* W,
                               u16* Wp, const float* g, const float* b,
                               const float* res, float* Y,
                               int B, int Cin, int Cout, int Hin, int Win,
                               int KH, int KW, int stride, int pad, int act, int clip)
{
    int Hout = (Hin + 2 * pad - KH) / stride + 1;
    int Wout = (Win + 2 * pad - KW) / stride + 1;
    int Nsp = Hout * Wout;
    dim3 grid((Nsp + BN - 1) / BN, (Cout + BM - 1) / BM, B);
    if ((Cin % BK) == 0 && !clip) {
        // pack weights to bf16 fragment-ordered tiles, then async-LDS conv
        int nMB = (Cout + BM - 1) / BM;
        dim3 rg(Cin / BK, KH * KW, nMB);
        repack_w_k<<<rg, 256, 0, st>>>(W, Wp, Cout, Cin, KH * KW);
        conv_wmma_s<<<grid, 256, 0, st>>>(X, Wp, g, b, res, Y, Cin, Cout, Hin, Win,
                                          Hout, Wout, KH, KW, stride, pad, act,
                                          kBNMul, magic40(Wout));
    } else {
        conv_wmma_g<<<grid, 256, 0, st>>>(X, W, g, b, res, Y, Cin, Cout, Hin, Win,
                                          Hout, Wout, KH, KW, stride, pad, act, clip,
                                          kBNMul, magic40(Wout), magic40(KH * KW),
                                          magic40(KW));
    }
}

struct BlockP {
    const float *w1, *w2, *w3, *g1, *b1, *g2, *b2, *g3, *b3, *wd, *gd, *bd;
    bool down;
};

extern "C" void kernel_launch(void* const* d_in, const int* in_sizes, int n_in,
                              void* d_out, int out_size, void* d_ws, size_t ws_size,
                              hipStream_t stream)
{
    (void)in_sizes; (void)n_in; (void)out_size; (void)ws_size;
    auto F = [&](int i) { return (const float*)d_in[i]; };

    // --- parameter map: jax pytree flatten order (sorted dict keys) ---
    const float *attn_b1 = F(0),  *attn_b2 = F(1),  *attn_w1 = F(2),  *attn_w2 = F(3);
    const float *fpn_b1  = F(4),  *fpn_b2  = F(5),  *fpn_w1  = F(6),  *fpn_w2  = F(7);
    const float *head_b1 = F(8),  *head_b2 = F(9),  *head_b3 = F(10);
    const float *head_w1 = F(11), *head_w2 = F(12), *head_w3 = F(13);

    int idx = 14;
    const int nblk[4] = {3, 4, 6, 3};
    BlockP blocks[16]; int nb = 0;
    for (int li = 0; li < 4; ++li) {
        for (int bi = 0; bi < nblk[li]; ++bi) {
            BlockP p{};
            if (bi == 0) { // keys: b1,b2,b3,bd,g1,g2,g3,gd,w1,w2,w3,wd
                p.b1 = F(idx + 0); p.b2 = F(idx + 1); p.b3 = F(idx + 2); p.bd = F(idx + 3);
                p.g1 = F(idx + 4); p.g2 = F(idx + 5); p.g3 = F(idx + 6); p.gd = F(idx + 7);
                p.w1 = F(idx + 8); p.w2 = F(idx + 9); p.w3 = F(idx + 10); p.wd = F(idx + 11);
                p.down = true; idx += 12;
            } else {       // keys: b1,b2,b3,g1,g2,g3,w1,w2,w3
                p.b1 = F(idx + 0); p.b2 = F(idx + 1); p.b3 = F(idx + 2);
                p.g1 = F(idx + 3); p.g2 = F(idx + 4); p.g3 = F(idx + 5);
                p.w1 = F(idx + 6); p.w2 = F(idx + 7); p.w3 = F(idx + 8);
                p.down = false; idx += 9;
            }
            blocks[nb++] = p;
        }
    }
    const float* stem_b = F(idx++);
    const float* stem_g = F(idx++);
    const float* stem_w = F(idx++);
    const float* x      = F(idx++);

    // --- workspace carve ---
    float* ws = (float*)d_ws;
    const size_t BIG = 12845056; // 4*256*112*112 == 4*64*224*224
    const size_t MID = 6422528;  // 4*128*112*112
    float* R0 = ws;
    float* R1 = ws + BIG;
    float* R2 = ws + 2 * BIG;        // residual / temps
    float* R3 = ws + 3 * BIG;        // bottleneck mid 1
    float* R4 = ws + 3 * BIG + MID;  // bottleneck mid 2
    u16*   WP = (u16*)(ws + 3 * BIG + 2 * MID); // packed weights (<= 19 MB)

    const int B = 4;

    // --- stem: clip -> conv7x7 s2 -> BN+relu -> maxpool3x3 s2 ---
    launch_conv(stream, x, stem_w, WP, stem_g, stem_b, nullptr, R0,
                B, 3, 64, 448, 448, 7, 7, 2, 3, /*relu*/1, /*clip*/1);
    {
        int total = B * 64 * 112 * 112;
        maxpool_k<<<(total + 255) / 256, 256, 0, stream>>>(R0, R1, B * 64, 224, 224, 112, 112);
    }

    // --- ResNet layers ---
    float* cur = R1;
    float* nxt = R0;
    int H = 112, ic = 64;
    const int mcs[4] = {64, 128, 256, 512};
    const int ocs[4] = {256, 512, 1024, 2048};
    int bidx = 0;
    for (int li = 0; li < 4; ++li) {
        int mc = mcs[li], oc = ocs[li];
        int st0 = (li == 0) ? 1 : 2;
        for (int bi = 0; bi < nblk[li]; ++bi) {
            const BlockP& p = blocks[bidx++];
            int s  = (bi == 0) ? st0 : 1;
            int H2 = (bi == 0) ? ((H + 2 - 3) / s + 1) : H;

            launch_conv(stream, cur, p.w1, WP, p.g1, p.b1, nullptr, R3,
                        B, ic, mc, H, H, 1, 1, 1, 0, 1, 0);
            launch_conv(stream, R3, p.w2, WP, p.g2, p.b2, nullptr, R4,
                        B, mc, mc, H, H, 3, 3, s, 1, 1, 0);
            const float* resPtr;
            if (p.down) {
                launch_conv(stream, cur, p.wd, WP, p.gd, p.bd, nullptr, R2,
                            B, ic, oc, H, H, 1, 1, s, 0, 0, 0);
                resPtr = R2;
            } else {
                resPtr = cur;
            }
            launch_conv(stream, R4, p.w3, WP, p.g3, p.b3, resPtr, nxt,
                        B, mc, oc, H2, H2, 1, 1, 1, 0, 1, 0);

            float* t = cur; cur = nxt; nxt = t;
            ic = oc; H = H2;
        }
    }
    // cur: (4, 2048, 14, 14)

    // --- attention ---
    launch_conv(stream, cur, attn_w1, WP, nullptr, attn_b1, nullptr, R3,
                B, 2048, 256, 14, 14, 1, 1, 1, 0, 1, 0);
    launch_conv(stream, R3, attn_w2, WP, nullptr, attn_b2, nullptr, R2,
                B, 256, 2048, 14, 14, 1, 1, 1, 0, 3, 0);
    {
        size_t n = (size_t)B * 2048 * 14 * 14;
        mul_k<<<(unsigned)((n + 255) / 256), 256, 0, stream>>>(cur, R2, nxt, n);
        float* t = cur; cur = nxt; nxt = t;
    }

    // --- FPN ---
    launch_conv(stream, cur, fpn_w1, WP, nullptr, fpn_b1, nullptr, R3,
                B, 2048, 512, 14, 14, 3, 3, 1, 1, 2, 0);
    {
        int total = B * 512 * 28 * 28;
        upsample2_k<<<(total + 255) / 256, 256, 0, stream>>>(R3, R4, B * 512, 14, 14);
    }
    launch_conv(stream, R4, fpn_w2, WP, nullptr, fpn_b2, nullptr, R2,
                B, 512, 256, 28, 28, 3, 3, 1, 1, 2, 0);

    // --- head ---
    launch_conv(stream, R2, head_w1, WP, nullptr, head_b1, nullptr, R3,
                B, 256, 512, 28, 28, 3, 3, 1, 1, 2, 0);
    launch_conv(stream, R3, head_w2, WP, nullptr, head_b2, nullptr, R4,
                B, 512, 256, 28, 28, 3, 3, 1, 1, 2, 0);
    launch_conv(stream, R4, head_w3, WP, nullptr, head_b3, nullptr, R2,
                B, 256, 6, 28, 28, 1, 1, 1, 0, 0, 0);

    // --- decode + NMS into d_out: [boxes | obj | labels | keep] ---
    const int HW = 28 * 28;
    float* out_boxes = (float*)d_out;
    float* out_obj   = out_boxes + (size_t)B * HW * 4;
    float* out_lab   = out_obj   + (size_t)B * HW;
    float* out_keep  = out_lab   + (size_t)B * HW;
    {
        int total = B * HW;
        decode_k<<<(total + 255) / 256, 256, 0, stream>>>(R2, out_boxes, out_obj,
                                                          out_lab, B, 6, HW);
    }
    nms_k<<<B, 256, 0, stream>>>(out_boxes, out_obj, out_keep, HW, 0.5f);
}
